// Pct_73203422593028
// MI455X (gfx1250) — compile-verified, matrix operand order checked by reading, not run on hardware
//
#include <hip/hip_runtime.h>

#define NPTS 32768
#define CDIM 128
#define KNBR 16
#define NODES_PER_WAVE 8

typedef __attribute__((ext_vector_type(16))) _Float16 v16h;
typedef __attribute__((ext_vector_type(8)))  float    v8f;
typedef __attribute__((ext_vector_type(4)))  unsigned int u32x4;
typedef __attribute__((ext_vector_type(8)))  int      i32x8;
typedef __attribute__((ext_vector_type(4)))  int      i32x4;

// ---------------------------------------------------------------------------
// LDS staging of a 32KB (16384 x f16) pre-transposed weight tile.
// Preferred path: Tensor Data Mover (tensor_load_to_lds), one DMA per block,
// descriptor per CDNA5 ISA:
//   group0: count=1, lds_addr, global_addr[56:0], type=2
//   group1: data_size=1(2B), tensor_dim0=tile_dim0=16384, tensor_dim1=tile_dim1=1,
//           tensor_dim0_stride=16384, workgroup_mask=0 (not in a cluster)
// This toolchain exposes the 6-arg builtin (extra int32x8 group, then cpol).
// Fallback: plain dword copy loop.
// ---------------------------------------------------------------------------
#if defined(__AMDGCN__) && __has_builtin(__builtin_amdgcn_tensor_load_to_lds)
#define HAVE_TDM 1
#else
#define HAVE_TDM 0
#endif

__device__ __forceinline__ void lds_stage_32kb(void* lds_dst,
                                               const _Float16* __restrict__ gsrc,
                                               int tid, int nthreads) {
#if HAVE_TDM
  if (tid == 0) {
    unsigned long long ga = (unsigned long long)(const void*)gsrc;
    unsigned lo = (unsigned)(unsigned long long)lds_dst;  // addr[31:0] = LDS offset
    u32x4 g0 = { 0x1u,                                  // count=1, user mode
                 lo,                                    // lds_addr
                 (unsigned)ga,                          // global_addr[31:0]
                 (unsigned)((ga >> 32) & 0x01FFFFFFull) | 0x80000000u };  // addr[56:32] | type=2
    i32x8 g1 = { 0x00010000,        // workgroup_mask=0, data_size=1 (2 bytes)
                 0x40000000,        // tensor_dim0[15:0]=16384 << 16
                 0x00010000,        // tensor_dim0[31:16]=0 | tensor_dim1[15:0]=1
                 0x40000000,        // tensor_dim1[31:16]=0 | tile_dim0=16384
                 0x00000001,        // tile_dim1=1 | tile_dim2=0
                 16384,             // tensor_dim0_stride[31:0]
                 0, 0 };            // stride0[47:32], stride1
    i32x4 gz4 = { 0, 0, 0, 0 };
    i32x8 gz8 = { 0, 0, 0, 0, 0, 0, 0, 0 };
    __builtin_amdgcn_tensor_load_to_lds(g0, g1, gz4, gz4, gz8, 0);
  }
#else
  const int* src = (const int*)gsrc;
  int* dst = (int*)lds_dst;
  for (int i = tid; i < 8192; i += nthreads) dst[i] = src[i];
#endif
}

__device__ __forceinline__ void lds_stage_wait() {
#if defined(__AMDGCN__) && __has_builtin(__builtin_amdgcn_s_wait_tensorcnt)
  __builtin_amdgcn_s_wait_tensorcnt(0);   // waves that issued nothing pass immediately
#endif
  __syncthreads();
}

// ---------------------------------------------------------------------------
// A-fragment loader: 16-bit A 16x32 layout (CDNA5 ISA):
//   lanes 0-15 : M = lane,    K = {0..7} in elems 0-7, K = {16..23} in elems 8-15
//   lanes 16-31: M = lane-16, K = {8..15},              K = {24..31}
// ---------------------------------------------------------------------------
__device__ __forceinline__ v16h load_a_f32(const float* __restrict__ row, int kc, int khi) {
  const float* p0 = row + kc * 32 + khi;
  const float* p1 = p0 + 16;
  v16h a;
#pragma unroll
  for (int e = 0; e < 8; ++e) {
    a[e]     = (_Float16)p0[e];
    a[8 + e] = (_Float16)p1[e];
  }
  return a;
}

// B-fragment from LDS; weights stored transposed Bt[n][k] so reads are contiguous.
__device__ __forceinline__ v16h load_b_lds(const _Float16* Bt, int col0, int n, int kc, int khi) {
  const _Float16* p = Bt + (col0 + n) * CDIM + kc * 32 + khi;
  v16h b;
#pragma unroll
  for (int e = 0; e < 8; ++e) {
    b[e]     = p[e];
    b[8 + e] = p[16 + e];
  }
  return b;
}

// ---------------------------------------------------------------------------
// Kernel 0: convert a 128x128 f32 weight matrix into transposed f16 (Out[n][k]).
// ---------------------------------------------------------------------------
__global__ void wcvt_kernel(const float* __restrict__ W, _Float16* __restrict__ Out) {
  int i = blockIdx.x * 256 + threadIdx.x;
  int k = i >> 7, n = i & 127;
  Out[n * CDIM + k] = (_Float16)W[k * CDIM + n];
}

// ---------------------------------------------------------------------------
// Kernel 1: brute-force KNN, one thread per query, coords tiled through LDS.
// coords layout: [N][4] = (0, x, y, z). Keeps sorted top-16 in registers.
// ---------------------------------------------------------------------------
__global__ void knn_kernel(const float* __restrict__ coords, int* __restrict__ idx_out) {
  __shared__ float4 tile[256];
  const float4* c4 = (const float4*)coords;
  int q = blockIdx.x * 256 + threadIdx.x;
  float4 qc = c4[q];

  float bd[KNBR];
  int   bi[KNBR];
#pragma unroll
  for (int i = 0; i < KNBR; ++i) { bd[i] = 3.0e38f; bi[i] = 0; }

  for (int t = 0; t < NPTS; t += 256) {
    __syncthreads();
    tile[threadIdx.x] = c4[t + threadIdx.x];
    __syncthreads();
#pragma unroll 4
    for (int j = 0; j < 256; ++j) {
      float4 p = tile[j];
      float dx = qc.y - p.y, dy = qc.z - p.z, dz = qc.w - p.w, dw = qc.x - p.x;
      float d = dx * dx + dy * dy + dz * dz + dw * dw;
      if (d < bd[KNBR - 1]) {
        bd[KNBR - 1] = d;
        bi[KNBR - 1] = t + j;
#pragma unroll
        for (int s = KNBR - 1; s > 0; --s) {
          if (bd[s] < bd[s - 1]) {
            float td = bd[s]; bd[s] = bd[s - 1]; bd[s - 1] = td;
            int   ti = bi[s]; bi[s] = bi[s - 1]; bi[s - 1] = ti;
          }
        }
      }
    }
  }
#pragma unroll
  for (int i = 0; i < KNBR; ++i) idx_out[q * KNBR + i] = bi[i];
}

// ---------------------------------------------------------------------------
// Kernel 2: Y[N,128] = X[N,128] @ W[128,128] + bias   (WMMA f16->f32)
// Wt is pre-transposed f16; staged to LDS via TDM. Block = 4 waves, 128 rows.
// ---------------------------------------------------------------------------
__global__ void gemm_bias_kernel(const float* __restrict__ X,
                                 const _Float16* __restrict__ Wt,
                                 const float* __restrict__ bias,
                                 float* __restrict__ Y) {
  __shared__ _Float16 Bt[CDIM * CDIM];  // 32 KB
  lds_stage_32kb(Bt, Wt, threadIdx.x, blockDim.x);
  lds_stage_wait();

  int wave = threadIdx.x >> 5;
  int lane = threadIdx.x & 31;
  int m    = lane & 15;
  int khi  = (lane >> 4) * 8;
  int rbase = (lane >> 4) * 8;  // C/D rows: lanes 0-15 -> M=r, lanes 16-31 -> M=r+8

  for (int rt = 0; rt < 8; ++rt) {
    int row0 = (blockIdx.x * 8 + rt) * 16;
    const float* rowptr = X + (size_t)(row0 + m) * CDIM;
    v16h afr[4];
#pragma unroll
    for (int kc = 0; kc < 4; ++kc) afr[kc] = load_a_f32(rowptr, kc, khi);

#pragma unroll
    for (int cc = 0; cc < 2; ++cc) {
      int col0 = (wave * 2 + cc) * 16;
      v8f acc = {};
#pragma unroll
      for (int kc = 0; kc < 4; ++kc) {
        v16h b = load_b_lds(Bt, col0, m, kc, khi);
        acc = __builtin_amdgcn_wmma_f32_16x16x32_f16(false, afr[kc], false, b,
                                                     (short)0, acc, false, false);
      }
      float bv = bias[col0 + m];
#pragma unroll
      for (int r = 0; r < 8; ++r)
        Y[(size_t)(row0 + rbase + r) * CDIM + col0 + m] = acc[r] + bv;
    }
  }
}

// ---------------------------------------------------------------------------
// Kernel 3: fused neighborhood attention, one wave per node.
//   nf[m,:] = src[idx[q,m],:] + (xyz_q - xyz_j) @ pos_w + pos_b  (A-fragments)
//   scores  = (nf @ kw + kb) . Q[q]/sqrt(C); softmax over 16 neighbors
//   out[q]  = softmax . (nf @ vw + vb)
// kwT16/vwT16 are pre-transposed f16 weights staged to LDS via TDM.
// ---------------------------------------------------------------------------
__global__ void attn_kernel(const float* __restrict__ src,
                            const float* __restrict__ Qbuf,
                            const float* __restrict__ coords,
                            const int*   __restrict__ idx,
                            const float* __restrict__ posw,
                            const float* __restrict__ posb,
                            const _Float16* __restrict__ kwT16,
                            const float* __restrict__ kb,
                            const _Float16* __restrict__ vwT16,
                            const float* __restrict__ vb,
                            float* __restrict__ out) {
  __shared__ _Float16 kwT[CDIM * CDIM];   // 32 KB
  __shared__ _Float16 vwT[CDIM * CDIM];   // 32 KB
  __shared__ float s_posw[3 * CDIM];
  __shared__ float s_posb[CDIM];
  __shared__ float s_kb[CDIM];
  __shared__ float s_vb[CDIM];

  lds_stage_32kb(kwT, kwT16, threadIdx.x, blockDim.x);
  lds_stage_32kb(vwT, vwT16, threadIdx.x, blockDim.x);
  for (int i = threadIdx.x; i < 3 * CDIM; i += blockDim.x) s_posw[i] = posw[i];
  for (int i = threadIdx.x; i < CDIM; i += blockDim.x) {
    s_posb[i] = posb[i]; s_kb[i] = kb[i]; s_vb[i] = vb[i];
  }
  lds_stage_wait();

  const float4* c4 = (const float4*)coords;
  int wave = threadIdx.x >> 5;
  int lane = threadIdx.x & 31;
  int m    = lane & 15;            // neighbor row AND output column within a tile
  int khi  = (lane >> 4) * 8;
  bool lo  = (lane < 16);
  const float scale = 0.08838834764831845f;  // 1/sqrt(128)

  for (int iter = 0; iter < NODES_PER_WAVE; ++iter) {
    int q = (blockIdx.x * 4 + wave) * NODES_PER_WAVE + iter;
    int j = idx[q * KNBR + m];
    float4 pq = c4[q];
    float4 pj = c4[j];
    float dx = pq.y - pj.y, dy = pq.z - pj.z, dz = pq.w - pj.w;

    // Build A-fragments of nf = src[j] + posenc (f32 -> f16)
    const float* sr = src + (size_t)j * CDIM;
    v16h afr[4];
#pragma unroll
    for (int kc = 0; kc < 4; ++kc) {
      v16h a;
#pragma unroll
      for (int e = 0; e < 8; ++e) {
        int k1 = kc * 32 + khi + e;
        int k2 = k1 + 16;
        float v1 = sr[k1] + dx * s_posw[k1] + dy * s_posw[CDIM + k1] +
                   dz * s_posw[2 * CDIM + k1] + s_posb[k1];
        float v2 = sr[k2] + dx * s_posw[k2] + dy * s_posw[CDIM + k2] +
                   dz * s_posw[2 * CDIM + k2] + s_posb[k2];
        a[e]     = (_Float16)v1;
        a[8 + e] = (_Float16)v2;
      }
      afr[kc] = a;
    }

    // Per-lane slice of the query vector: column n = m of each of 8 tiles
    float Qv[8];
#pragma unroll
    for (int t = 0; t < 8; ++t) Qv[t] = Qbuf[(size_t)q * CDIM + t * 16 + m];

    // K projection via WMMA + partial scores
    float sc[8];
#pragma unroll
    for (int r = 0; r < 8; ++r) sc[r] = 0.f;
#pragma unroll
    for (int t = 0; t < 8; ++t) {
      v8f acc = {};
#pragma unroll
      for (int kc = 0; kc < 4; ++kc) {
        v16h b = load_b_lds(kwT, t * 16, m, kc, khi);
        acc = __builtin_amdgcn_wmma_f32_16x16x32_f16(false, afr[kc], false, b,
                                                     (short)0, acc, false, false);
      }
      float qv = Qv[t];
      float kbv = s_kb[t * 16 + m];
#pragma unroll
      for (int r = 0; r < 8; ++r) sc[r] += (acc[r] + kbv) * qv;
    }

    // Reduce over the 16 column-lanes of each half-wave
#pragma unroll
    for (int r = 0; r < 8; ++r) {
      sc[r] += __shfl_xor(sc[r], 1, 32);
      sc[r] += __shfl_xor(sc[r], 2, 32);
      sc[r] += __shfl_xor(sc[r], 4, 32);
      sc[r] += __shfl_xor(sc[r], 8, 32);
    }
    // Exchange halves: lower half holds rows 0-7, upper rows 8-15
    float sLo[8], sHi[8];
#pragma unroll
    for (int r = 0; r < 8; ++r) {
      float oth = __shfl_xor(sc[r], 16, 32);
      sLo[r] = (lo ? sc[r] : oth) * scale;
      sHi[r] = (lo ? oth : sc[r]) * scale;
    }
    // Softmax over all 16 (computed redundantly in every lane, constant-indexed)
    float mx = -3.0e38f;
#pragma unroll
    for (int r = 0; r < 8; ++r) mx = fmaxf(mx, fmaxf(sLo[r], sHi[r]));
    float awLo[8], awHi[8], ssum = 0.f;
#pragma unroll
    for (int r = 0; r < 8; ++r) {
      awLo[r] = __expf(sLo[r] - mx);
      awHi[r] = __expf(sHi[r] - mx);
      ssum += awLo[r] + awHi[r];
    }
    float inv = 1.f / ssum;
#pragma unroll
    for (int r = 0; r < 8; ++r) { awLo[r] *= inv; awHi[r] *= inv; }

    // V projection via WMMA + softmax-weighted column sums
    float po[8];
#pragma unroll
    for (int t = 0; t < 8; ++t) {
      v8f acc = {};
#pragma unroll
      for (int kc = 0; kc < 4; ++kc) {
        v16h b = load_b_lds(vwT, t * 16, m, kc, khi);
        acc = __builtin_amdgcn_wmma_f32_16x16x32_f16(false, afr[kc], false, b,
                                                     (short)0, acc, false, false);
      }
      float vbv = s_vb[t * 16 + m];
      float s = 0.f;
#pragma unroll
      for (int r = 0; r < 8; ++r) {
        float w = lo ? awLo[r] : awHi[r];  // lanes 16-31 hold rows 8-15
        s += w * (acc[r] + vbv);
      }
      po[t] = s;
    }
#pragma unroll
    for (int t = 0; t < 8; ++t) po[t] += __shfl_xor(po[t], 16, 32);
    if (lo) {
#pragma unroll
      for (int t = 0; t < 8; ++t) out[(size_t)q * CDIM + t * 16 + m] = po[t];
    }
  }
}

// ---------------------------------------------------------------------------
// Kernel 4: out = LayerNorm(x + y; eps = 128.0) * g + b   (one block per row)
// ---------------------------------------------------------------------------
__global__ void add_ln_kernel(const float* __restrict__ x,
                              const float* __restrict__ y,
                              const float* __restrict__ g,
                              const float* __restrict__ b,
                              float* __restrict__ out) {
  __shared__ float red[8];
  size_t base = (size_t)blockIdx.x * CDIM;
  int c = threadIdx.x;
  int lane = c & 31, w = c >> 5;

  float v = x[base + c] + y[base + c];
  float s = v;
  s += __shfl_xor(s, 1, 32); s += __shfl_xor(s, 2, 32);
  s += __shfl_xor(s, 4, 32); s += __shfl_xor(s, 8, 32);
  s += __shfl_xor(s, 16, 32);
  if (lane == 0) red[w] = s;
  __syncthreads();
  float mean = (red[0] + red[1] + red[2] + red[3]) * (1.0f / CDIM);
  float d = v - mean;
  float s2 = d * d;
  s2 += __shfl_xor(s2, 1, 32); s2 += __shfl_xor(s2, 2, 32);
  s2 += __shfl_xor(s2, 4, 32); s2 += __shfl_xor(s2, 8, 32);
  s2 += __shfl_xor(s2, 16, 32);
  if (lane == 0) red[4 + w] = s2;
  __syncthreads();
  float var = (red[4] + red[5] + red[6] + red[7]) * (1.0f / CDIM);
  out[base + c] = d * rsqrtf(var + 128.0f) * g[c] + b[c];  // LN_EPS = float(C)
}

// ---------------------------------------------------------------------------
extern "C" void kernel_launch(void* const* d_in, const int* in_sizes, int n_in,
                              void* d_out, int out_size, void* d_ws, size_t ws_size,
                              hipStream_t stream) {
  const float* coords = (const float*)d_in[0];
  const float* feats  = (const float*)d_in[1];
  const float* pos_w  = (const float*)d_in[2];
  const float* pos_b  = (const float*)d_in[3];
  const float* pos1_w = (const float*)d_in[4];
  const float* pos1_b = (const float*)d_in[5];
  const float* qw0 = (const float*)d_in[6];  const float* qb0 = (const float*)d_in[7];
  const float* kw0 = (const float*)d_in[8];  const float* kb0 = (const float*)d_in[9];
  const float* vw0 = (const float*)d_in[10]; const float* vb0 = (const float*)d_in[11];
  const float* qw1 = (const float*)d_in[12]; const float* qb1 = (const float*)d_in[13];
  const float* kw1 = (const float*)d_in[14]; const float* kb1 = (const float*)d_in[15];
  const float* vw1 = (const float*)d_in[16]; const float* vb1 = (const float*)d_in[17];
  const float* lin_w = (const float*)d_in[18]; const float* lin_b = (const float*)d_in[19];
  const float* g0  = (const float*)d_in[20]; const float* be0 = (const float*)d_in[21];
  const float* g1  = (const float*)d_in[22]; const float* be1 = (const float*)d_in[23];

  char* ws = (char*)d_ws;
  const size_t MB = (size_t)1 << 20;
  int*   idx = (int*)ws;                       //  2 MB (N*16 int)
  float* Qb  = (float*)(ws + 2 * MB);          // 16 MB (N*128 f32)
  float* O0  = (float*)(ws + 18 * MB);         // 16 MB
  float* O1  = (float*)(ws + 34 * MB);         // 16 MB
  float* H   = (float*)(ws + 50 * MB);         // 16 MB
  _Float16* W16 = (_Float16*)(ws + 66 * MB);   // 7 x 32 KB transposed f16 weights
  float* H1  = Qb;                             // reuse Q buffer after attention 2

  _Float16* qw0T = W16 + 0 * 16384;
  _Float16* kw0T = W16 + 1 * 16384;
  _Float16* vw0T = W16 + 2 * 16384;
  _Float16* qw1T = W16 + 3 * 16384;
  _Float16* kw1T = W16 + 4 * 16384;
  _Float16* vw1T = W16 + 5 * 16384;
  _Float16* linT = W16 + 6 * 16384;

  // 0. pre-transpose + f16-convert all GEMM weights (TDM-friendly layout)
  wcvt_kernel<<<64, 256, 0, stream>>>(qw0,  qw0T);
  wcvt_kernel<<<64, 256, 0, stream>>>(kw0,  kw0T);
  wcvt_kernel<<<64, 256, 0, stream>>>(vw0,  vw0T);
  wcvt_kernel<<<64, 256, 0, stream>>>(qw1,  qw1T);
  wcvt_kernel<<<64, 256, 0, stream>>>(kw1,  kw1T);
  wcvt_kernel<<<64, 256, 0, stream>>>(vw1,  vw1T);
  wcvt_kernel<<<64, 256, 0, stream>>>(lin_w, linT);

  // 1. KNN indices
  knn_kernel<<<NPTS / 256, 256, 0, stream>>>(coords, idx);

  // 2. Attention layer 0
  gemm_bias_kernel<<<NPTS / 128, 128, 0, stream>>>(feats, qw0T, qb0, Qb);
  attn_kernel<<<NPTS / (4 * NODES_PER_WAVE), 128, 0, stream>>>(
      feats, Qb, coords, idx, pos_w, pos_b, kw0T, kb0, vw0T, vb0, O0);

  // 3. Attention layer 1
  gemm_bias_kernel<<<NPTS / 128, 128, 0, stream>>>(O0, qw1T, qb1, Qb);
  attn_kernel<<<NPTS / (4 * NODES_PER_WAVE), 128, 0, stream>>>(
      O0, Qb, coords, idx, pos1_w, pos1_b, kw1T, kb1, vw1T, vb1, O1);

  // 4. h = LN(feats + o1); h1 = h @ lin_w + lin_b; out = LN(h + h1)
  add_ln_kernel<<<NPTS, 128, 0, stream>>>(feats, O1, g0, be0, H);
  gemm_bias_kernel<<<NPTS / 128, 128, 0, stream>>>(H, linT, lin_b, H1);
  add_ln_kernel<<<NPTS, 128, 0, stream>>>(H, H1, g1, be1, (float*)d_out);
}
